// TimeConvAttentionBlock1_36704790512041
// MI455X (gfx1250) — matrix operand order repaired and checked by reading, not compiled, not run on hardware
//
#include <hip/hip_runtime.h>
#include <hip/hip_bf16.h>

#define DI __device__ __forceinline__

typedef __attribute__((ext_vector_type(16))) __bf16 v16bf;
typedef __attribute__((ext_vector_type(8)))  float  v8f;
typedef int v4i __attribute__((vector_size(16)));

union FragA { v16bf v; unsigned int u[8]; unsigned short s[16]; };

constexpr int BB  = 16;
constexpr int C   = 96;
constexpr int HH  = 128;
constexpr int WW  = 128;
constexpr int HW  = HH * WW;            // 16384
constexpr int NTOT = BB * C * HW;       // 25165824
constexpr int ASTR = 56;                // A-tile LDS/global row stride (bf16 elems), 112B = 7x16B
constexpr int BSTR = 34;                // B-tile LDS row stride

DI unsigned short f2bf(float f) {
  unsigned int u = __float_as_uint(f);
  unsigned int r = (u + 0x7FFFu + ((u >> 16) & 1u)) >> 16;
  return (unsigned short)r;
}
DI float bf2f(unsigned short h) {
  return __uint_as_float(((unsigned int)h) << 16);
}
DI float silu_f(float x) { return x / (1.0f + expf(-x)); }

// --------------------------------------------------------------------------
// CDNA5 async global->LDS copy (ASYNCcnt) with compile-safe fallback.
// ROCm 7.2 signature: (v4i __device__* src, v4i __shared__* dst, imm, imm)
// --------------------------------------------------------------------------
DI void async_b128(void* lds_dst, const void* gsrc) {
#if defined(__gfx1250__) && __has_builtin(__builtin_amdgcn_global_load_async_to_lds_b128)
  __builtin_amdgcn_global_load_async_to_lds_b128(
      (__attribute__((address_space(1))) v4i*)gsrc,
      (__attribute__((address_space(3))) v4i*)lds_dst, 0, 0);
#else
  *(uint4*)lds_dst = *(const uint4*)gsrc;
#endif
}
DI void wait_async_all() {
#if defined(__gfx1250__) && __has_builtin(__builtin_amdgcn_s_wait_asynccnt)
  __builtin_amdgcn_s_wait_asynccnt(0);
#endif
}

// ---------------------------------------------------------------------------
// WMMA tile body: block computes M=96 (6 m-tiles) x N=64 (4 n-tiles), 8 waves,
// 3 (mi,ni) tiles per wave. LDS layouts: As[co][kk] stride 56, Bs[n][kk] stride
// 34; K-pairs contiguous so each fragment is 8x ds_load_b32 (fused to 2addr).
// A-frag (16x32 bf16): lane half=l>>4, m=l&15; vgpr r -> K = (r&4?16:0)+half*8+(r&3)*2
// B-frag (32x16 bf16): lane n=l&15, half=l>>4; vgpr r -> K = (r&4?16:0)+half*8+(r&3)*2
// ---------------------------------------------------------------------------
#define WMMA_STEP(AsArr, BsArr)                                                   \
  do {                                                                            \
    _Pragma("unroll")                                                             \
    for (int j = 0; j < 3; ++j) {                                                 \
      int p_ = wv * 3 + j;                                                        \
      int mi_ = p_ >> 2, ni_ = p_ & 3;                                            \
      FragA fa_, fb_;                                                             \
      int coL_ = mi_ * 16 + nl;                                                   \
      _Pragma("unroll")                                                           \
      for (int r = 0; r < 8; ++r) {                                               \
        int kb_ = ((r & 4) << 2) + half * 8 + (r & 3) * 2;                        \
        fa_.u[r] = *(const unsigned int*)&AsArr[coL_ * ASTR + kb_];               \
      }                                                                           \
      int nn_ = ni_ * 16 + nl;                                                    \
      _Pragma("unroll")                                                           \
      for (int r = 0; r < 8; ++r) {                                               \
        fb_.u[r] = *(const unsigned int*)&BsArr[nn_ * BSTR + ((r & 4) << 2) + half * 8 + (r & 3) * 2];  \
      }                                                                           \
      acc[j] = __builtin_amdgcn_wmma_f32_16x16x32_bf16(                           \
          false, fa_.v, false, fb_.v, (short)0, acc[j], false, false);            \
    }                                                                             \
  } while (0)

// Stage one k-step A-tile (96 rows x 112B) via async copies: 672 b128 chunks.
#define STAGE_A_ASYNC(AsArr, srcBase)                                             \
  do {                                                                            \
    const unsigned short* ws_ = (srcBase);                                        \
    for (int i_ = t; i_ < 96 * 7; i_ += 256) {                                    \
      int row_ = i_ / 7, ch_ = (i_ % 7) * 8;                                      \
      async_b128(&AsArr[row_ * ASTR + ch_], ws_ + row_ * ASTR + ch_);             \
    }                                                                             \
  } while (0)

// ---------------------------------------------------------------------------
// conv3x3 implicit GEMM: M=96 (all Cout), K = 9 taps * 96 Cin (27 steps of 32),
// N-tile = 64 consecutive x in one row. Epilogue: +bias, store f32, GN stats.
// wp layout: [kstep=27][co=96][ASTR] bf16 (pre-packed, async-friendly).
// ---------------------------------------------------------------------------
__global__ __launch_bounds__(256)
void k_conv3x3(const unsigned short* __restrict__ xin,   // bf16 NCHW
               const unsigned short* __restrict__ wp,    // bf16 [27][96][ASTR]
               const float* __restrict__ bias,
               float* __restrict__ yout,                 // f32 NCHW (raw conv+bias)
               float* __restrict__ stats)                // [B*4][2] atomic accum
{
  __shared__ __align__(16) unsigned short As[96 * ASTR];
  __shared__ __align__(16) unsigned short Bs[64 * BSTR];
  __shared__ float sg[4], sq[4];

  const int t    = threadIdx.x;
  const int b    = blockIdx.y;
  const int tile = blockIdx.x;            // 0..255 : y = tile>>1, xseg = (tile&1)*64
  const int y    = tile >> 1;
  const int x0   = (tile & 1) * 64;
  const int wv   = t >> 5, lane = t & 31;
  const int half = lane >> 4, nl = lane & 15;

  v8f acc[3];
#pragma unroll
  for (int j = 0; j < 3; ++j)
#pragma unroll
    for (int r = 0; r < 8; ++r) acc[j][r] = 0.0f;

  for (int s = 0; s < 27; ++s) {
    const int tap = s / 3;
    const int dy = tap / 3 - 1, dx = tap % 3 - 1;
    const int cb = (s % 3) * 32;
    __syncthreads();
    // A: weights, contiguous in packed layout -> async global->LDS
    STAGE_A_ASYNC(As, wp + (size_t)s * 96 * ASTR);
    // B: im2col gather (transposing into [n][kk]) via VALU path
    for (int i = t; i < 32 * 64; i += 256) {
      int kk = i >> 6, n = i & 63;
      int yy = y + dy;
      int xx = x0 + n + dx;
      unsigned short vv = 0;
      if (yy >= 0 && yy < HH && xx >= 0 && xx < WW)
        vv = xin[((b * C + cb + kk) * HH + yy) * WW + xx];
      Bs[n * BSTR + kk] = vv;
    }
    // prefetch next step's im2col rows (one cacheline per kk row)
    if (s + 1 < 27 && t < 32) {
      int s2 = s + 1;
      int tap2 = s2 / 3;
      int dy2 = tap2 / 3 - 1, dx2 = tap2 % 3 - 1;
      int cb2 = (s2 % 3) * 32;
      int yy = y + dy2;
      int xx = x0 + dx2;
      if (yy >= 0 && yy < HH && xx >= 0 && xx < WW)
        __builtin_prefetch(&xin[((b * C + cb2 + t) * HH + yy) * WW + xx], 0, 1);
    }
    wait_async_all();
    __syncthreads();
    WMMA_STEP(As, Bs);
  }

  if (t < 4) { sg[t] = 0.0f; sq[t] = 0.0f; }
  __syncthreads();

  float lg[4] = {0, 0, 0, 0}, lq[4] = {0, 0, 0, 0};
#pragma unroll
  for (int j = 0; j < 3; ++j) {
    int p = wv * 3 + j;
    int mi = p >> 2, ni = p & 3;
    int pos = y * WW + x0 + ni * 16 + nl;
#pragma unroll
    for (int r = 0; r < 8; ++r) {
      int co = mi * 16 + half * 8 + r;
      float v = acc[j][r] + bias[co];
      yout[(b * C + co) * HW + pos] = v;
      int gi = co / 24;
      lg[gi] += v;
      lq[gi] += v * v;
    }
  }
#pragma unroll
  for (int gi = 0; gi < 4; ++gi) {
    atomicAdd(&sg[gi], lg[gi]);
    atomicAdd(&sq[gi], lq[gi]);
  }
  __syncthreads();
  if (t < 8) {
    int gi = t >> 1;
    float v = (t & 1) ? sq[gi] : sg[gi];
    atomicAdd(&stats[(b * 4 + gi) * 2 + (t & 1)], v);
  }
}

// ---------------------------------------------------------------------------
// qkv 1x1 conv as GEMM: M=288 (3 groups of 96 via blockIdx.y), K=96, N=64/block.
// wq layout [kstep=3][co=288][ASTR] bf16. Output bf16 [B][288][HW].
// ---------------------------------------------------------------------------
__global__ __launch_bounds__(256)
void k_gemm_qkv(const unsigned short* __restrict__ hn,   // bf16 [B][96][HW]
                const unsigned short* __restrict__ wq,   // bf16 [3][288][ASTR]
                const float* __restrict__ bq,
                unsigned short* __restrict__ qkv)        // bf16 [B][288][HW]
{
  __shared__ __align__(16) unsigned short As[96 * ASTR];
  __shared__ __align__(16) unsigned short Bs[64 * BSTR];
  const int t  = threadIdx.x;
  const int mg = blockIdx.y;
  const int b  = blockIdx.z;
  const int np = blockIdx.x * 64;
  const int wv = t >> 5, lane = t & 31, half = lane >> 4, nl = lane & 15;

  v8f acc[3];
#pragma unroll
  for (int j = 0; j < 3; ++j)
#pragma unroll
    for (int r = 0; r < 8; ++r) acc[j][r] = 0.0f;

  for (int s = 0; s < 3; ++s) {
    int cb = s * 32;
    __syncthreads();
    STAGE_A_ASYNC(As, wq + (size_t)(s * 288 + mg * 96) * ASTR);
    for (int i = t; i < 32 * 64; i += 256) {
      int kk = i >> 6, n = i & 63;
      Bs[n * BSTR + kk] = hn[(b * C + cb + kk) * HW + np + n];
    }
    wait_async_all();
    __syncthreads();
    WMMA_STEP(As, Bs);
  }

#pragma unroll
  for (int j = 0; j < 3; ++j) {
    int p = wv * 3 + j;
    int mi = p >> 2, ni = p & 3;
    int pos = np + ni * 16 + nl;
#pragma unroll
    for (int r = 0; r < 8; ++r) {
      int c = mg * 96 + mi * 16 + half * 8 + r;
      qkv[(b * 288 + c) * HW + pos] = f2bf(acc[j][r] + bq[c]);
    }
  }
}

// ---------------------------------------------------------------------------
// proj 1x1 GEMM: M=96, K=96, input abuf f32 (converted in staging), epilogue
// adds bias + residual h1, outputs bf16 (conv2 input).
// ---------------------------------------------------------------------------
__global__ __launch_bounds__(256)
void k_gemm_proj(const float* __restrict__ abuf,          // f32 [B][96][HW]
                 const unsigned short* __restrict__ wpj,  // bf16 [3][96][ASTR]
                 const float* __restrict__ bpj,
                 const float* __restrict__ resid,         // h1 f32
                 unsigned short* __restrict__ h2)         // bf16 [B][96][HW]
{
  __shared__ __align__(16) unsigned short As[96 * ASTR];
  __shared__ __align__(16) unsigned short Bs[64 * BSTR];
  const int t  = threadIdx.x;
  const int b  = blockIdx.z;
  const int np = blockIdx.x * 64;
  const int wv = t >> 5, lane = t & 31, half = lane >> 4, nl = lane & 15;

  v8f acc[3];
#pragma unroll
  for (int j = 0; j < 3; ++j)
#pragma unroll
    for (int r = 0; r < 8; ++r) acc[j][r] = 0.0f;

  for (int s = 0; s < 3; ++s) {
    int cb = s * 32;
    __syncthreads();
    STAGE_A_ASYNC(As, wpj + (size_t)(s * 96) * ASTR);
    for (int i = t; i < 32 * 64; i += 256) {
      int kk = i >> 6, n = i & 63;
      Bs[n * BSTR + kk] = f2bf(abuf[(b * C + cb + kk) * HW + np + n]);
    }
    wait_async_all();
    __syncthreads();
    WMMA_STEP(As, Bs);
  }

#pragma unroll
  for (int j = 0; j < 3; ++j) {
    int p = wv * 3 + j;
    int mi = p >> 2, ni = p & 3;
    int pos = np + ni * 16 + nl;
#pragma unroll
    for (int r = 0; r < 8; ++r) {
      int c = mi * 16 + half * 8 + r;
      int idx = (b * C + c) * HW + pos;
      h2[idx] = f2bf(acc[j][r] + bpj[c] + resid[idx]);
    }
  }
}

// ---------------------------------------------------------------------------
// GN1 + SiLU + FiLM, accumulates GN2 stats. One block = (b, c, 2048-pos seg).
// ---------------------------------------------------------------------------
__global__ __launch_bounds__(256)
void k_gn1_film(const float* __restrict__ y1, const float* __restrict__ st1,
                const float* __restrict__ g1, const float* __restrict__ be1,
                const float* __restrict__ tvec,
                float* __restrict__ h1, float* __restrict__ st2)
{
  __shared__ float ssum, ssq;
  int bx = blockIdx.x;
  int b = bx / 768, rem = bx % 768, c = rem >> 3, seg = rem & 7;
  int gi = c / 24;
  const float invN = 1.0f / (24.0f * (float)HW);
  float mu = st1[(b * 4 + gi) * 2] * invN;
  float va = st1[(b * 4 + gi) * 2 + 1] * invN - mu * mu;
  float rs = rsqrtf(va + 1e-5f);
  float gg = g1[c], bb = be1[c];
  float sh = tvec[b * 192 + c], bi = tvec[b * 192 + 96 + c];
  int base = (b * C + c) * HW + seg * 2048;
  if (threadIdx.x == 0) { ssum = 0.0f; ssq = 0.0f; }
  __syncthreads();
  float ls = 0.0f, lq = 0.0f;
#pragma unroll
  for (int i = 0; i < 8; ++i) {
    int idx = base + threadIdx.x + i * 256;
    float v = y1[idx];
    float xn = (v - mu) * rs * gg + bb;
    float h = silu_f(xn) * (1.0f + sh) + bi;
    h1[idx] = h;
    ls += h;
    lq += h * h;
  }
  atomicAdd(&ssum, ls);
  atomicAdd(&ssq, lq);
  __syncthreads();
  if (threadIdx.x == 0) {
    atomicAdd(&st2[(b * 4 + gi) * 2], ssum);
    atomicAdd(&st2[(b * 4 + gi) * 2 + 1], ssq);
  }
}

__global__ __launch_bounds__(256)
void k_gn2_apply(const float* __restrict__ h1, const float* __restrict__ st,
                 const float* __restrict__ ga, const float* __restrict__ ba,
                 unsigned short* __restrict__ outp)       // bf16
{
  int bx = blockIdx.x;
  int b = bx / 768, rem = bx % 768, c = rem >> 3, seg = rem & 7;
  int gi = c / 24;
  const float invN = 1.0f / (24.0f * (float)HW);
  float mu = st[(b * 4 + gi) * 2] * invN;
  float va = st[(b * 4 + gi) * 2 + 1] * invN - mu * mu;
  float rs = rsqrtf(va + 1e-5f);
  float gg = ga[c], bb = ba[c];
  int base = (b * C + c) * HW + seg * 2048;
#pragma unroll
  for (int i = 0; i < 8; ++i) {
    int idx = base + threadIdx.x + i * 256;
    outp[idx] = f2bf((h1[idx] - mu) * rs * gg + bb);
  }
}

__global__ __launch_bounds__(256)
void k_gn3_out(const float* __restrict__ y2, const float* __restrict__ st,
               const float* __restrict__ g2, const float* __restrict__ be2,
               float* __restrict__ outp)
{
  int bx = blockIdx.x;
  int b = bx / 768, rem = bx % 768, c = rem >> 3, seg = rem & 7;
  int gi = c / 24;
  const float invN = 1.0f / (24.0f * (float)HW);
  float mu = st[(b * 4 + gi) * 2] * invN;
  float va = st[(b * 4 + gi) * 2 + 1] * invN - mu * mu;
  float rs = rsqrtf(va + 1e-5f);
  float gg = g2[c], bb = be2[c];
  int base = (b * C + c) * HW + seg * 2048;
#pragma unroll
  for (int i = 0; i < 8; ++i) {
    int idx = base + threadIdx.x + i * 256;
    outp[idx] = silu_f((y2[idx] - mu) * rs * gg + bb);
  }
}

// ---------------------------------------------------------------------------
// Channel "window attention": per (b, window, head): attn = softmax(q k^T / sqrt(6))
// over HEAD_DIM (6x6), contracted over L=64. One block of 64 threads per
// window-head; thread t = spatial position l within window. shift folds the roll.
// ---------------------------------------------------------------------------
__global__ __launch_bounds__(64)
void k_attn(const unsigned short* __restrict__ qkv, float* __restrict__ abuf,
            int shift, int accumulate)
{
  __shared__ float prod[36 * 65];
  __shared__ float pm[36];
  const int t   = threadIdx.x;
  const int win = blockIdx.x;
  const int h   = blockIdx.y;
  const int b   = blockIdx.z;
  const int wy = win >> 4, wx = win & 15;
  const int ly = t >> 3, lx = t & 7;
  const int y = (wy * 8 + shift + ly) & 127;
  const int x = (wx * 8 + shift + lx) & 127;
  const int sp = y * 128 + x;

  float q[6], k[6], v[6];
  const unsigned short* base = qkv + (size_t)b * 288 * HW;
#pragma unroll
  for (int d = 0; d < 6; ++d) {
    q[d] = bf2f(base[(h * 6 + d) * HW + sp]);
    k[d] = bf2f(base[(96 + h * 6 + d) * HW + sp]);
    v[d] = bf2f(base[(192 + h * 6 + d) * HW + sp]);
  }
#pragma unroll
  for (int d = 0; d < 6; ++d)
#pragma unroll
    for (int e = 0; e < 6; ++e) prod[(d * 6 + e) * 65 + t] = q[d] * k[e];
  __syncthreads();
  if (t < 36) {
    float s = 0.0f;
    for (int i = 0; i < 64; ++i) s += prod[t * 65 + i];
    pm[t] = s * 0.40824829046386302f;   // 6^-0.5
  }
  __syncthreads();
  if (t < 6) {
    float mx = -1e30f;
    for (int e = 0; e < 6; ++e) mx = fmaxf(mx, pm[t * 6 + e]);
    float ex[6], sum = 0.0f;
    for (int e = 0; e < 6; ++e) { ex[e] = expf(pm[t * 6 + e] - mx); sum += ex[e]; }
    float inv = 1.0f / sum;
    for (int e = 0; e < 6; ++e) pm[t * 6 + e] = ex[e] * inv;
  }
  __syncthreads();
  float* ob = abuf + (size_t)b * C * HW;
#pragma unroll
  for (int d = 0; d < 6; ++d) {
    float o = 0.0f;
#pragma unroll
    for (int e = 0; e < 6; ++e) o += pm[d * 6 + e] * v[e];
    o *= 0.5f;
    float* p = &ob[(h * 6 + d) * HW + sp];
    if (accumulate) *p += o; else *p = o;
  }
}

// ---------------------------------------------------------------------------
// small helpers: packing / conversion / time-MLP
// ---------------------------------------------------------------------------
__global__ void k_cvt_bf16(const float* __restrict__ in, unsigned short* __restrict__ outp, int n)
{
  int i = blockIdx.x * blockDim.x + threadIdx.x;
  int stride = gridDim.x * blockDim.x;
  for (; i < n; i += stride) outp[i] = f2bf(in[i]);
}

// conv weights (co,ci,3,3) -> packed [kstep=27][co][ASTR], k-order = tap*96+ci
__global__ void k_pack_w3(const float* __restrict__ w, unsigned short* __restrict__ wp)
{
  int i = blockIdx.x * 256 + threadIdx.x;       // 96 * 864
  if (i >= 96 * 864) return;
  int co = i / 864, kodr = i % 864;
  int tap = kodr / 96, ci = kodr % 96;
  int s = kodr >> 5, kk = kodr & 31;
  wp[(s * 96 + co) * ASTR + kk] = f2bf(w[(co * 96 + ci) * 9 + tap]);
}

// 1x1 weights (M,96,1,1) -> packed [kstep=3][co=M][ASTR]
__global__ void k_pack_w1x1(const float* __restrict__ w, unsigned short* __restrict__ wp, int M)
{
  int i = blockIdx.x * 256 + threadIdx.x;       // M*96
  if (i >= M * 96) return;
  int co = i / 96, ci = i % 96;
  int s = ci >> 5, kk = ci & 31;
  wp[(s * M + co) * ASTR + kk] = f2bf(w[co * 96 + ci]);
}

__global__ void k_tmlp(const float* __restrict__ te, const float* __restrict__ wt,
                       const float* __restrict__ bt, float* __restrict__ tvec)
{
  int b = blockIdx.x;        // 16
  int o = threadIdx.x;       // 192
  float s = bt[o];
  for (int kk = 0; kk < 256; ++kk) s += te[b * 256 + kk] * wt[o * 256 + kk];
  tvec[b * 192 + o] = silu_f(s);
}

// ---------------------------------------------------------------------------
extern "C" void kernel_launch(void* const* d_in, const int* in_sizes, int n_in,
                              void* d_out, int out_size, void* d_ws, size_t ws_size,
                              hipStream_t stream)
{
  (void)in_sizes; (void)n_in; (void)out_size; (void)ws_size;
  const float* x      = (const float*)d_in[0];
  const float* t_emb  = (const float*)d_in[1];
  const float* w1     = (const float*)d_in[2];
  const float* b1     = (const float*)d_in[3];
  const float* g1     = (const float*)d_in[4];
  const float* be1    = (const float*)d_in[5];
  const float* wt     = (const float*)d_in[6];
  const float* bt     = (const float*)d_in[7];
  const float* qkv_w  = (const float*)d_in[8];
  const float* qkv_b  = (const float*)d_in[9];
  const float* proj_w = (const float*)d_in[10];
  const float* proj_b = (const float*)d_in[11];
  const float* ga     = (const float*)d_in[12];
  const float* ba     = (const float*)d_in[13];
  const float* w2     = (const float*)d_in[14];
  const float* b2     = (const float*)d_in[15];
  const float* g2     = (const float*)d_in[16];
  const float* be2    = (const float*)d_in[17];
  float* out = (float*)d_out;

  char* ws = (char*)d_ws;
  const size_t SZF = (size_t)NTOT * 4;   // 100,663,296 B
  const size_t SZH = (size_t)NTOT * 2;   //  50,331,648 B

  float*          r0   = (float*)(ws);                              // y1 / abuf / y2
  float*          h1   = (float*)(ws + SZF);                        // residual
  unsigned short* qkv  = (unsigned short*)(ws + 2 * SZF);           // bf16, 3*SZH
  unsigned short* ubuf = (unsigned short*)(ws + 2 * SZF + 3 * SZH); // xbf / hn / h2
  char* sm = ws + 2 * SZF + 4 * SZH;
  unsigned short* w1pak = (unsigned short*)(sm);            // 27*96*56*2 = 290304 B
  unsigned short* w2pak = (unsigned short*)(sm + 290304);   // 290304 B
  unsigned short* wqpak = (unsigned short*)(sm + 580608);   // 3*288*56*2 = 96768 B
  unsigned short* wppak = (unsigned short*)(sm + 677376);   // 3*96*56*2 = 32256 B
  float* tvec = (float*)(sm + 709632);                      // 12288 B
  float* st1  = (float*)(sm + 721920);                      // 3*128 floats
  float* st2  = st1 + 128;
  float* st3  = st2 + 128;

  (void)hipMemsetAsync(st1, 0, 3 * 128 * sizeof(float), stream);

  // prep
  k_cvt_bf16<<<12288, 256, 0, stream>>>(x, ubuf, NTOT);
  k_pack_w3<<<(96 * 864 + 255) / 256, 256, 0, stream>>>(w1, w1pak);
  k_pack_w3<<<(96 * 864 + 255) / 256, 256, 0, stream>>>(w2, w2pak);
  k_pack_w1x1<<<(288 * 96 + 255) / 256, 256, 0, stream>>>(qkv_w, wqpak, 288);
  k_pack_w1x1<<<(96 * 96 + 255) / 256, 256, 0, stream>>>(proj_w, wppak, 96);
  k_tmlp<<<16, 192, 0, stream>>>(t_emb, wt, bt, tvec);

  // conv1 + GN1 stats
  k_conv3x3<<<dim3(256, 16), 256, 0, stream>>>(ubuf, w1pak, b1, r0, st1);
  // GN1 + SiLU + FiLM -> h1 (residual), GN2 stats
  k_gn1_film<<<12288, 256, 0, stream>>>(r0, st1, g1, be1, tvec, h1, st2);
  // GN2 apply -> bf16 (ubuf reused)
  k_gn2_apply<<<12288, 256, 0, stream>>>(h1, st2, ga, ba, ubuf);
  // qkv GEMM
  k_gemm_qkv<<<dim3(256, 3, 16), 256, 0, stream>>>(ubuf, wqpak, qkv_b, qkv);
  // two window passes (roll folded into index shift), averaged into r0
  k_attn<<<dim3(256, 16, 16), 64, 0, stream>>>(qkv, r0, 0, 0);
  k_attn<<<dim3(256, 16, 16), 64, 0, stream>>>(qkv, r0, 4, 1);
  // proj GEMM + residual -> bf16 (ubuf reused as h2)
  k_gemm_proj<<<dim3(256, 1, 16), 256, 0, stream>>>(r0, wppak, proj_b, h1, ubuf);
  // conv2 + GN3 stats (r0 reused as y2)
  k_conv3x3<<<dim3(256, 16), 256, 0, stream>>>(ubuf, w2pak, b2, r0, st3);
  // GN3 + SiLU -> out
  k_gn3_out<<<12288, 256, 0, stream>>>(r0, st3, g2, be2, out);
}
